// SDKTModel_11871289606608
// MI455X (gfx1250) — matrix operand order repaired and checked by reading, not compiled, not run on hardware
//
#include <hip/hip_runtime.h>
#include <hip/hip_bf16.h>
#include <math.h>

typedef __attribute__((ext_vector_type(16))) _Float16     v16h;
typedef __attribute__((ext_vector_type(8)))  _Float16     v8h;
typedef __attribute__((ext_vector_type(8)))  float        v8f;
typedef __attribute__((ext_vector_type(4)))  unsigned int v4u;
typedef __attribute__((ext_vector_type(8)))  unsigned int v8u;

#define B_   256
#define T_   256
#define H_   512
#define G_   2048
#define NBLK 64
#define NTHR 256
#define KCMAX 1536   // enc LSTM: 512 (h) + 1024 (x)

// ---------------- device helpers ----------------

__device__ __forceinline__ float sigf(float x) { return 1.f / (1.f + expf(-x)); }

// A-matrix 16x32 f16 fragment (ISA 7.12.2): lane holds row M=lane&15;
// lanes 0-15: K {k0..k0+7, k0+16..k0+23}; lanes 16-31: +8.
__device__ __forceinline__ v16h load_a16(const _Float16* row_k0, int lane) {
    const int koff = (lane & 16) ? 8 : 0;
    union { v16h v; v8h h[2]; } u;
    u.h[0] = *(const v8h*)(row_k0 + koff);
    u.h[1] = *(const v8h*)(row_k0 + 16 + koff);
    return u.v;
}

// B-matrix 32x16 f16 fragment: lane holds column N=lane&15;
// lanes 0-15: K {k0..k0+7, k0+16..k0+23}; lanes 16-31: +8.
__device__ __forceinline__ v16h load_b16(const _Float16* col_k0, int lane) {
    const int koff = (lane & 16) ? 8 : 0;
    union { v16h v; v8h h[2]; } u;
    u.h[0] = *(const v8h*)(col_k0 + koff);
    u.h[1] = *(const v8h*)(col_k0 + 16 + koff);
    return u.v;
}

// Grid-wide barrier: monotonic counter zeroed before each persistent launch.
__device__ __forceinline__ void grid_sync(unsigned* bar, unsigned target) {
    __threadfence();
    __syncthreads();
    if (threadIdx.x == 0) {
        atomicAdd(bar, 1u);
        while (__hip_atomic_load(bar, __ATOMIC_RELAXED, __HIP_MEMORY_SCOPE_AGENT) < target) {
            __builtin_amdgcn_s_sleep(2);
        }
    }
    __syncthreads();
    __threadfence();
}

// ---------------- small prep kernels ----------------

__global__ void k_pack(_Float16* __restrict__ Wc, const float* __restrict__ whh,
                       const float* __restrict__ wih, int Kx) {
    const int Kc = H_ + Kx;
    size_t total = (size_t)G_ * Kc;
    size_t i = (size_t)blockIdx.x * blockDim.x + threadIdx.x;
    if (i >= total) return;
    int k = (int)(i % Kc);
    size_t g = i / Kc;
    float v = (k < H_) ? whh[g * H_ + k] : wih[g * Kx + (k - H_)];
    Wc[i] = (_Float16)v;
}

__global__ void k_bias(float* __restrict__ o, const float* __restrict__ a,
                       const float* __restrict__ b) {
    int i = blockIdx.x * blockDim.x + threadIdx.x;
    if (i < G_) o[i] = a[i] + b[i];
}

__global__ void k_cvt(_Float16* __restrict__ o, const float* __restrict__ s, int n) {
    int i = blockIdx.x * blockDim.x + threadIdx.x;
    if (i < n) o[i] = (_Float16)s[i];
}

__global__ void k_embed_enc(const int* __restrict__ q, const int* __restrict__ a,
                            const int* __restrict__ p, const int* __restrict__ f,
                            const float* __restrict__ tw, const float* __restrict__ aw,
                            const float* __restrict__ pw, const float* __restrict__ fw,
                            _Float16* __restrict__ out) {
    size_t i = (size_t)blockIdx.x * blockDim.x + threadIdx.x;
    if (i >= (size_t)B_ * T_ * 288) return;
    int col = (int)(i % 288);
    size_t m = i / 288;
    float v;
    if (col < 128)      v = tw[(size_t)q[m] * 128 + col];
    else if (col < 256) v = aw[(size_t)a[m] * 128 + (col - 128)];
    else if (col < 272) v = pw[(size_t)p[m] * 16 + (col - 256)];
    else                v = fw[(size_t)f[m] * 16 + (col - 272)];
    out[i] = (_Float16)v;
}

__global__ void k_embed_dec(const int* __restrict__ q, const int* __restrict__ a,
                            const int* __restrict__ p, const int* __restrict__ f,
                            const int* __restrict__ lastA,
                            const float* __restrict__ tw, const float* __restrict__ aw,
                            const float* __restrict__ pw, const float* __restrict__ fw,
                            _Float16* __restrict__ decx, _Float16* __restrict__ predin) {
    size_t i = (size_t)blockIdx.x * blockDim.x + threadIdx.x;
    if (i >= (size_t)B_ * T_ * 288) return;
    int col = (int)(i % 288);
    size_t m = i / 288;
    int b = (int)(m / T_), t = (int)(m % T_);
    float v;
    if (col < 128)      v = tw[(size_t)q[m] * 128 + col];
    else if (col < 144) v = pw[(size_t)p[m] * 16 + (col - 128)];
    else if (col < 160) v = fw[(size_t)f[m] * 16 + (col - 144)];
    else {
        int pa = (t == 0) ? lastA[b] : a[m - 1];
        v = aw[(size_t)pa * 128 + (col - 160)];
    }
    decx[m * 288 + col] = (_Float16)v;
    if (col < 160) predin[m * 672 + col] = (_Float16)v;
}

// ---------------- persistent fused LSTM ----------------
// Weight-stationary: each block's 64-column slice of Wc (64 x Kc f16, <=192KB)
// is DMA'd once into LDS by the Tensor Data Mover, then reused for all 256
// timesteps.  Per step: gbuf = [h16 | x16_t] @ Wc^T + bias, grid barrier,
// elementwise LSTM gates, grid barrier.
__global__ void __launch_bounds__(NTHR, 1)
k_lstm(const _Float16* __restrict__ x16, int ldx,
       const _Float16* __restrict__ Wc, int Kc,
       const float* __restrict__ bias,
       _Float16* __restrict__ h16, float* __restrict__ cbuf,
       float* __restrict__ gbuf,
       _Float16* __restrict__ dest, int ldd, int dcol, int trev,
       unsigned* __restrict__ bar) {
    __shared__ __align__(16) _Float16 wlds[64 * KCMAX];

    const int lane  = threadIdx.x & 31;
    const int wv    = threadIdx.x >> 5;
    const int ntile = blockIdx.x >> 1;                      // block-uniform
    const int n0    = ntile * 64;
    const int m0    = (((blockIdx.x & 1) << 3) + wv) * 16;  // 16 row tiles over B=256
    const int ncol  = lane & 15;
    const int hi8   = (lane & 16) ? 8 : 0;
    const int bb    = m0 + (lane & 15);                     // batch row for A loads
    const int nkx   = (Kc - H_) >> 5;
    const int tidg  = blockIdx.x * NTHR + threadIdx.x;
    unsigned target = 0;

    // ---- TDM: stage this block's 64 x Kc weight slice into LDS ----
    if (wv == 0) {
        const unsigned lds_base = (unsigned)(size_t)&wlds[0];
        const unsigned long long ga =
            (unsigned long long)(size_t)(Wc + (size_t)n0 * Kc);
        v4u g0;
        g0[0] = 1u;                                   // count=1 (valid), user mode
        g0[1] = lds_base;                             // lds_addr
        g0[2] = (unsigned)ga;                         // global_addr[31:0]
        g0[3] = ((unsigned)(ga >> 32) & 0x01FFFFFFu)  // global_addr[56:32]
                | (2u << 30);                         // type=2 (image)
        v8u g1;
        g1[0] = 1u << 16;                             // data_size=2 bytes
        g1[1] = ((unsigned)Kc & 0xFFFFu) << 16;       // tensor_dim0[15:0]
        g1[2] = ((unsigned)Kc >> 16) | (64u << 16);   // tensor_dim0[31:16] | tensor_dim1
        g1[3] = ((unsigned)Kc) << 16;                 // tile_dim0 = Kc
        g1[4] = 64u;                                  // tile_dim1 = 64 rows
        g1[5] = (unsigned)Kc;                         // tensor_dim0_stride[31:0]
        g1[6] = 0u;
        g1[7] = 0u;
        asm volatile("tensor_load_to_lds %0, %1" :: "s"(g0), "s"(g1) : "memory");
        __builtin_amdgcn_s_wait_tensorcnt(0);
    }
    __syncthreads();

    const _Float16* hrow = h16 + (size_t)bb * H_;

    for (int t = 0; t < T_; ++t) {
        const int tx = trev ? (T_ - 1 - t) : t;
        const _Float16* xrow = x16 + (size_t)(bb * T_ + tx) * ldx;
        if (t + 1 < T_) {
            const int txn = trev ? (T_ - 2 - t) : (t + 1);
            __builtin_prefetch(x16 + (size_t)(bb * T_ + txn) * ldx, 0, 3);
        }

        v8f acc[4];
        #pragma unroll
        for (int j = 0; j < 4; ++j) {
            float bv = bias[n0 + j * 16 + ncol];
            #pragma unroll
            for (int r = 0; r < 8; ++r) acc[j][r] = bv;
        }

        // recurrent part: K in [0, 512), A = h16, B from LDS
        for (int ks = 0; ks < H_ / 32; ++ks) {
            const int k0 = ks << 5;
            v16h am = load_a16(hrow + k0, lane);
            #pragma unroll
            for (int j = 0; j < 4; ++j) {
                v16h bm = load_b16(&wlds[(size_t)(j * 16 + ncol) * Kc + k0], lane);
                acc[j] = __builtin_amdgcn_wmma_f32_16x16x32_f16(
                    false, am, false, bm, (short)0, acc[j], false, false);
            }
        }
        // input part: K in [512, Kc), A = x_t, B from LDS
        for (int ks = 0; ks < nkx; ++ks) {
            const int k0 = ks << 5;
            v16h am = load_a16(xrow + k0, lane);
            #pragma unroll
            for (int j = 0; j < 4; ++j) {
                v16h bm = load_b16(&wlds[(size_t)(j * 16 + ncol) * Kc + H_ + k0], lane);
                acc[j] = __builtin_amdgcn_wmma_f32_16x16x32_f16(
                    false, am, false, bm, (short)0, acc[j], false, false);
            }
        }
        #pragma unroll
        for (int j = 0; j < 4; ++j)
            #pragma unroll
            for (int r = 0; r < 8; ++r)
                gbuf[(size_t)(m0 + r + hi8) * G_ + (n0 + j * 16 + ncol)] = acc[j][r];

        target += NBLK; grid_sync(bar, target);

        // gates: i,f,g,o at columns hh, 512+hh, 1024+hh, 1536+hh
        for (int e = tidg; e < B_ * H_; e += NBLK * NTHR) {
            int b = e >> 9, hh = e & (H_ - 1);
            const float* gb = gbuf + (size_t)b * G_;
            float gi = gb[hh], gf = gb[H_ + hh], gg = gb[2 * H_ + hh], go = gb[3 * H_ + hh];
            float cn = sigf(gf) * cbuf[e] + sigf(gi) * tanhf(gg);
            float hn = sigf(go) * tanhf(cn);
            cbuf[e] = cn;
            h16[e] = (_Float16)hn;
            if (dest) dest[(size_t)(b * T_ + tx) * ldd + dcol + hh] = (_Float16)hn;
        }
        target += NBLK; grid_sync(bar, target);
    }
}

// ---------------- prediction head ----------------
__global__ void k_pred1(const _Float16* __restrict__ pin, const _Float16* __restrict__ w1,
                        const float* __restrict__ b1, _Float16* __restrict__ hid) {
    const int lane = threadIdx.x & 31;
    const int wid  = blockIdx.x * 8 + (threadIdx.x >> 5);
    const int n0   = (wid & 7) * 64;     // 512/64
    const int m0   = (wid >> 3) * 16;    // 65536/16
    const int ncol = lane & 15;
    const int hi8  = (lane & 16) ? 8 : 0;

    v8f acc[4];
    #pragma unroll
    for (int j = 0; j < 4; ++j) {
        float bv = b1[n0 + j * 16 + ncol];
        #pragma unroll
        for (int r = 0; r < 8; ++r) acc[j][r] = bv;
    }
    const _Float16* arow = pin + (size_t)(m0 + (lane & 15)) * 672;
    for (int ks = 0; ks < 21; ++ks) {
        const int k0 = ks << 5;
        v16h am = load_a16(arow + k0, lane);
        #pragma unroll
        for (int j = 0; j < 4; ++j) {
            const _Float16* bcol = w1 + (size_t)(n0 + j * 16 + ncol) * 672 + k0;
            v16h bm = load_b16(bcol, lane);
            acc[j] = __builtin_amdgcn_wmma_f32_16x16x32_f16(
                false, am, false, bm, (short)0, acc[j], false, false);
        }
    }
    #pragma unroll
    for (int j = 0; j < 4; ++j)
        #pragma unroll
        for (int r = 0; r < 8; ++r)
            hid[(size_t)(m0 + r + hi8) * H_ + (n0 + j * 16 + ncol)] = (_Float16)tanhf(acc[j][r]);
}

__global__ void k_pred2(const _Float16* __restrict__ hid, const float* __restrict__ w2,
                        const float* __restrict__ b2, float* __restrict__ out) {
    const int lane = threadIdx.x & 31;
    const int m = blockIdx.x * 8 + (threadIdx.x >> 5);
    const _Float16* row = hid + (size_t)m * H_;
    float s = 0.f;
    for (int i = lane; i < H_; i += 32) s += (float)row[i] * w2[i];
    #pragma unroll
    for (int off = 16; off; off >>= 1) s += __shfl_xor(s, off, 32);
    if (lane == 0) out[m] = 1.f / (1.f + expf(-(s + b2[0])));
}

// ---------------- host side ----------------

static inline void* carve(char*& p, size_t bytes) {
    void* r = (void*)p;
    p += (bytes + 255) & ~(size_t)255;
    return r;
}

extern "C" void kernel_launch(void* const* d_in, const int* in_sizes, int n_in,
                              void* d_out, int out_size, void* d_ws, size_t ws_size,
                              hipStream_t stream) {
    (void)in_sizes; (void)n_in; (void)out_size; (void)ws_size;
    const int*   enc_q   = (const int*)d_in[0];
    const int*   enc_a   = (const int*)d_in[1];
    const int*   enc_pos = (const int*)d_in[2];
    const int*   enc_fmt = (const int*)d_in[3];
    const int*   dec_q   = (const int*)d_in[4];
    const int*   dec_a   = (const int*)d_in[5];
    const int*   dec_pos = (const int*)d_in[6];
    const int*   dec_fmt = (const int*)d_in[7];
    const int*   lastA   = (const int*)d_in[8];
    const float* t_emb   = (const float*)d_in[9];
    const float* a_emb   = (const float*)d_in[10];
    const float* p_emb   = (const float*)d_in[11];
    const float* f_emb   = (const float*)d_in[12];
    const float* bif_wih = (const float*)d_in[13];
    const float* bif_whh = (const float*)d_in[14];
    const float* bif_bih = (const float*)d_in[15];
    const float* bif_bhh = (const float*)d_in[16];
    const float* bib_wih = (const float*)d_in[17];
    const float* bib_whh = (const float*)d_in[18];
    const float* bib_bih = (const float*)d_in[19];
    const float* bib_bhh = (const float*)d_in[20];
    const float* enc_wih = (const float*)d_in[21];
    const float* enc_whh = (const float*)d_in[22];
    const float* enc_bih = (const float*)d_in[23];
    const float* enc_bhh = (const float*)d_in[24];
    const float* dec_wih = (const float*)d_in[25];
    const float* dec_whh = (const float*)d_in[26];
    const float* dec_bih = (const float*)d_in[27];
    const float* dec_bhh = (const float*)d_in[28];
    const float* pred_w1 = (const float*)d_in[29];
    const float* pred_b1 = (const float*)d_in[30];
    const float* pred_w2 = (const float*)d_in[31];
    const float* pred_b2 = (const float*)d_in[32];
    float* out = (float*)d_out;

    const size_t BT = (size_t)B_ * T_;
    char* p = (char*)d_ws;
    unsigned* bars   = (unsigned*)carve(p, 256);
    _Float16* h16_f  = (_Float16*)carve(p, (size_t)B_ * H_ * 2);
    _Float16* h16_b  = (_Float16*)carve(p, (size_t)B_ * H_ * 2);
    _Float16* h16_e  = (_Float16*)carve(p, (size_t)B_ * H_ * 2);
    float*    c_f    = (float*)   carve(p, (size_t)B_ * H_ * 4);
    float*    c_b    = (float*)   carve(p, (size_t)B_ * H_ * 4);
    float*    c_e    = (float*)   carve(p, (size_t)B_ * H_ * 4);
    size_t state_bytes = (size_t)p - (size_t)bars;
    _Float16* encx16 = (_Float16*)carve(p, BT * 288 * 2);
    _Float16* decx16 = (_Float16*)carve(p, BT * 288 * 2);
    _Float16* bi16   = (_Float16*)carve(p, BT * 1024 * 2);
    _Float16* pin16  = (_Float16*)carve(p, BT * 672 * 2);
    _Float16* hid16  = (_Float16*)carve(p, BT * 512 * 2);
    _Float16* Wc_f   = (_Float16*)carve(p, (size_t)G_ * 800 * 2);
    _Float16* Wc_b   = (_Float16*)carve(p, (size_t)G_ * 800 * 2);
    _Float16* Wc_e   = (_Float16*)carve(p, (size_t)G_ * 1536 * 2);
    _Float16* Wc_d   = (_Float16*)carve(p, (size_t)G_ * 800 * 2);
    _Float16* w1_16  = (_Float16*)carve(p, (size_t)512 * 672 * 2);
    float*    bias_f = (float*)carve(p, G_ * 4);
    float*    bias_b = (float*)carve(p, G_ * 4);
    float*    bias_e = (float*)carve(p, G_ * 4);
    float*    bias_d = (float*)carve(p, G_ * 4);
    float*    gbuf   = (float*)carve(p, (size_t)B_ * G_ * 4);

    hipMemsetAsync(bars, 0, state_bytes, stream);   // counters + h0/c0 = 0

    {
        int g800  = (int)(((size_t)G_ * 800  + 255) / 256);
        int g1536 = (int)(((size_t)G_ * 1536 + 255) / 256);
        k_pack<<<g800,  256, 0, stream>>>(Wc_f, bif_whh, bif_wih, 288);
        k_pack<<<g800,  256, 0, stream>>>(Wc_b, bib_whh, bib_wih, 288);
        k_pack<<<g1536, 256, 0, stream>>>(Wc_e, enc_whh, enc_wih, 1024);
        k_pack<<<g800,  256, 0, stream>>>(Wc_d, dec_whh, dec_wih, 288);
        k_bias<<<G_ / 256, 256, 0, stream>>>(bias_f, bif_bih, bif_bhh);
        k_bias<<<G_ / 256, 256, 0, stream>>>(bias_b, bib_bih, bib_bhh);
        k_bias<<<G_ / 256, 256, 0, stream>>>(bias_e, enc_bih, enc_bhh);
        k_bias<<<G_ / 256, 256, 0, stream>>>(bias_d, dec_bih, dec_bhh);
        k_cvt<<<(512 * 672 + 255) / 256, 256, 0, stream>>>(w1_16, pred_w1, 512 * 672);
    }

    {
        int ge = (int)((BT * 288 + 255) / 256);
        k_embed_enc<<<ge, 256, 0, stream>>>(enc_q, enc_a, enc_pos, enc_fmt,
                                            t_emb, a_emb, p_emb, f_emb, encx16);
        k_embed_dec<<<ge, 256, 0, stream>>>(dec_q, dec_a, dec_pos, dec_fmt, lastA,
                                            t_emb, a_emb, p_emb, f_emb, decx16, pin16);
    }

    k_lstm<<<NBLK, NTHR, 0, stream>>>(encx16, 288, Wc_f, 800, bias_f,
                                      h16_f, c_f, gbuf, bi16, 1024, 0,   0, bars + 0);
    k_lstm<<<NBLK, NTHR, 0, stream>>>(encx16, 288, Wc_b, 800, bias_b,
                                      h16_b, c_b, gbuf, bi16, 1024, 512, 1, bars + 1);
    k_lstm<<<NBLK, NTHR, 0, stream>>>(bi16, 1024, Wc_e, 1536, bias_e,
                                      h16_e, c_e, gbuf, (_Float16*)nullptr, 0, 0, 0, bars + 2);
    k_lstm<<<NBLK, NTHR, 0, stream>>>(decx16, 288, Wc_d, 800, bias_d,
                                      h16_e, c_e, gbuf, pin16, 672, 160, 0, bars + 3);

    k_pred1<<<(int)(BT * 512 / (16 * 64) / 8), 256, 0, stream>>>(pin16, w1_16, pred_b1, hid16);
    k_pred2<<<(int)(BT / 8), 256, 0, stream>>>(hid16, pred_w2, pred_b2, out);
}
